// GTEProgramClassification_27986006900857
// MI455X (gfx1250) — matrix operand order repaired and checked by reading, not compiled, not run on hardware
//
#include <hip/hip_runtime.h>

#define N_SRC 50000
#define N_DST 32000
#define MAX_DEG 16
#define N_SUB 4
#define D 128
#define N_CLASSES 104
#define T_STEPS 15
#define LN_EPS 1e-5f

typedef __attribute__((ext_vector_type(16))) __bf16 v16bf;
typedef __attribute__((ext_vector_type(8)))  float  v8f;

struct Q2 { uint4 a, b; };

__device__ inline v16bf q2_to_v16bf(uint4 a, uint4 b) {
  Q2 q{a, b};
  return __builtin_bit_cast(v16bf, q);
}

// ---------------------------------------------------------------------------
// Kernel 1: feat_src = mean over 4 subtoken embeddings; emit fp32 + bf16 copies
// ---------------------------------------------------------------------------
__global__ void embed_mean_kernel(const int* __restrict__ token_id,
                                  const float* __restrict__ emb,
                                  float* __restrict__ feat_f32,
                                  unsigned short* __restrict__ feat_bf16) {
  int s = blockIdx.x;          // source node
  int j = threadIdx.x;         // feature column 0..127
  int t0 = token_id[s * N_SUB + 0];
  int t1 = token_id[s * N_SUB + 1];
  int t2 = token_id[s * N_SUB + 2];
  int t3 = token_id[s * N_SUB + 3];
  float v = 0.25f * (emb[t0 * D + j] + emb[t1 * D + j] +
                     emb[t2 * D + j] + emb[t3 * D + j]);
  feat_f32[s * D + j] = v;
  __bf16 h = (__bf16)v;
  feat_bf16[s * D + j] = __builtin_bit_cast(unsigned short, h);
}

// ---------------------------------------------------------------------------
// Kernel 2: fp32 -> bf16 weight conversion (row-major [384][128] preserved)
// ---------------------------------------------------------------------------
__global__ void cvt_bf16_kernel(const float* __restrict__ src,
                                unsigned short* __restrict__ dst, int n) {
  int i = blockIdx.x * blockDim.x + threadIdx.x;
  if (i < n) {
    __bf16 h = (__bf16)src[i];
    dst[i] = __builtin_bit_cast(unsigned short, h);
  }
}

// ---------------------------------------------------------------------------
// CDNA5 async global->LDS copy (ASYNCcnt-tracked, CDNA5 ISA)
// INST_OFFSET is added to BOTH the LDS and global addresses, so one address
// pair covers both 16-byte halves of a 32-byte segment.
// ---------------------------------------------------------------------------
__device__ __forceinline__ void async_copy_b256(unsigned lds_off,
                                                const unsigned short* gsrc) {
  asm volatile(
      "global_load_async_to_lds_b128 %0, %1, off\n\t"
      "global_load_async_to_lds_b128 %0, %1, off offset:16"
      :
      : "v"(lds_off), "v"(gsrc)
      : "memory");
}

__device__ __forceinline__ void wait_asynccnt0() {
  asm volatile("s_wait_asynccnt 0" ::: "memory");
}

// ---------------------------------------------------------------------------
// WMMA fragment builders (CDNA5 wave32 layouts, CDNA5 ISA)
// ---------------------------------------------------------------------------

// A-matrix 16x32 bf16 from LDS bf16 buffer [16][128].
// lanes 0-15: M=lane, K = {0..7, 16..23}+base ; lanes 16-31: M=lane-16, K = {8..15, 24..31}+base
__device__ inline v16bf load_a_bf16(const unsigned short* buf, int base_k) {
  int lane = threadIdx.x & 31;
  int m = lane & 15;
  int off = (lane & 16) ? 8 : 0;
  const unsigned short* row = buf + m * D + base_k + off;
  uint4 lo = *(const uint4*)(row);        // K = base+off .. +7
  uint4 hi = *(const uint4*)(row + 16);   // K = base+off+16 .. +23
  return q2_to_v16bf(lo, hi);
}

// Same A layout, built from fp32 LDS buffer [16][128] with bf16 conversion (h state)
__device__ inline v16bf load_a_f32(const float* buf, int base_k) {
  int lane = threadIdx.x & 31;
  int m = lane & 15;
  int off = (lane & 16) ? 8 : 0;
  const float* row = buf + m * D + base_k + off;
  v16bf a;
#pragma unroll
  for (int i = 0; i < 8; ++i) a[i] = (__bf16)row[i];
#pragma unroll
  for (int i = 0; i < 8; ++i) a[8 + i] = (__bf16)row[16 + i];
  return a;
}

// B-matrix 32x16 bf16 = (W^T) tile, W row-major bf16 [384][128].
// lane holds column n = lane&15; lanes 0-15: K = base+{0..7, 16..23}, lanes 16-31: K = base+{8..15, 24..31}.
// B(k,n) = W[n][k] -> two runs of 8 contiguous bf16 = 2 x 16 contiguous bytes per lane.
__device__ inline v16bf load_b_w(const unsigned short* __restrict__ w,
                                 int col0, int base_k) {
  int lane = threadIdx.x & 31;
  int n = lane & 15;
  int koff = (lane & 16) ? 8 : 0;
  const unsigned short* p = w + (size_t)(col0 + n) * D + base_k + koff;
  uint4 lo = *(const uint4*)(p);
  uint4 hi = *(const uint4*)(p + 16);
  return q2_to_v16bf(lo, hi);
}

// ---------------------------------------------------------------------------
// Kernel 3: fused GRU + LayerNorm + classifier. One block = 16 dst nodes,
// 128 threads = 4 waves; wave w computes N-tiles [6w .. 6w+5] of xp / hp.
// Weight B-fragments (48 x v16bf = 384 VGPRs/lane) are preloaded into
// registers once per block; the step loop contains no global loads.
// Message gather for step t+1 runs asynchronously (ASYNCcnt) under the WMMA
// and gate phases of step t.
// ---------------------------------------------------------------------------
__global__ void __launch_bounds__(128)
gru_fused_kernel(const int* __restrict__ neigh_idx,
                 const int* __restrict__ deg,
                 const float* __restrict__ feat_f32,
                 const unsigned short* __restrict__ feat_bf16,
                 const unsigned short* __restrict__ wih_bf,
                 const unsigned short* __restrict__ whh_bf,
                 const float* __restrict__ b_ih,
                 const float* __restrict__ b_hh,
                 const float* __restrict__ gamma,
                 const float* __restrict__ beta,
                 const float* __restrict__ wc,
                 const float* __restrict__ bc,
                 float* __restrict__ out) {
  __shared__ __align__(16) unsigned short sXt[16 * D];   // staged bf16 messages, 4 KB
  __shared__ __align__(16) float sH[16 * D];             // hidden state fp32,   8 KB
  __shared__ __align__(16) float sXp[16 * 3 * D];        // x-projection,       24 KB
  __shared__ __align__(16) float sHp[16 * 3 * D];        // h-projection,       24 KB
  __shared__ float sMu[16];
  __shared__ float sRs[16];

  const int n0 = blockIdx.x * 16;
  const int tid = threadIdx.x;
  const int wave = tid >> 5;
  const int lane = tid & 31;
  const int nloc = lane & 15;
  const int mbase = (lane & 16) ? 8 : 0;   // C/D layout: VGPR v -> M = mbase + v

  // staging decomposition: thread -> (node, 32B segment)
  const int snode = tid >> 3;   // 0..15
  const int sseg = tid & 7;     // 8 x 32B segments per 256B row
  const unsigned xt_lds_off =
      (unsigned)(size_t)(void*)(sXt + snode * D + sseg * 16);

  // kick off async gather of step-0 messages
  {
    int src = neigh_idx[(n0 + snode) * MAX_DEG + 0];
    async_copy_b256(xt_lds_off, feat_bf16 + (size_t)src * D + sseg * 16);
  }

  // zero hidden state
  for (int i = tid; i < 16 * D; i += 128) sH[i] = 0.0f;

  // per-thread copy of the 16 degrees (L0-cached reads)
  int dg[16];
#pragma unroll
  for (int m = 0; m < 16; ++m) dg[m] = deg[n0 + m];

  // ---- preload this wave's weight B-fragments + biases into registers ----
  v16bf bih[6][4], bhh[6][4];
  float bxr[6], bhr[6];
#pragma unroll
  for (int i = 0; i < 6; ++i) {
    int col0 = (wave * 6 + i) * 16;
#pragma unroll
    for (int c = 0; c < 4; ++c) {
      bih[i][c] = load_b_w(wih_bf, col0, 32 * c);
      bhh[i][c] = load_b_w(whh_bf, col0, 32 * c);
    }
    bxr[i] = b_ih[col0 + nloc];
    bhr[i] = b_hh[col0 + nloc];
  }

  for (int t = 0; t < T_STEPS; ++t) {
    // async gather for step t (issued last iteration / pre-loop) must be done
    wait_asynccnt0();
    __syncthreads();   // X_t ready for all waves; h updates visible

    // ---- pull A fragments into registers ----
    v16bf ax[4], ah[4];
#pragma unroll
    for (int c = 0; c < 4; ++c) {
      ax[c] = load_a_bf16(sXt, 32 * c);
      ah[c] = load_a_f32(sH, 32 * c);
    }
    __syncthreads();   // all waves consumed sXt (DScnt flushed by barrier)

    // ---- overlap: async gather of step t+1 messages into sXt ----
    if (t + 1 < T_STEPS) {
      int src = neigh_idx[(n0 + snode) * MAX_DEG + (t + 1)];
      async_copy_b256(xt_lds_off, feat_bf16 + (size_t)src * D + sseg * 16);
    }

    // ---- WMMA phase: xp = X_t @ w_ih^T, hp = h @ w_hh^T (register-resident B) ----
#pragma unroll
    for (int i = 0; i < 6; ++i) {
      v8f accx = {};
      v8f acch = {};
#pragma unroll
      for (int c = 0; c < 4; ++c) {
        accx = __builtin_amdgcn_wmma_f32_16x16x32_bf16(
            false, ax[c], false, bih[i][c], (short)0, accx, false, false);
      }
#pragma unroll
      for (int c = 0; c < 4; ++c) {
        acch = __builtin_amdgcn_wmma_f32_16x16x32_bf16(
            false, ah[c], false, bhh[i][c], (short)0, acch, false, false);
      }
      int n = (wave * 6 + i) * 16 + nloc;
      float bx = bxr[i];
      float bh = bhr[i];
#pragma unroll
      for (int v = 0; v < 8; ++v) {
        sXp[(mbase + v) * (3 * D) + n] = accx[v] + bx;
        sHp[(mbase + v) * (3 * D) + n] = acch[v] + bh;
      }
    }
    __syncthreads();   // xp/hp ready

    // ---- gate phase (fp32), thread j handles column j of all 16 nodes ----
    {
      int j = tid;   // 0..127
#pragma unroll
      for (int m = 0; m < 16; ++m) {
        if (t < dg[m] - 1) {
          const float* xp = sXp + m * (3 * D);
          const float* hp = sHp + m * (3 * D);
          float r = 1.0f / (1.0f + __expf(-(xp[j] + hp[j])));
          float z = 1.0f / (1.0f + __expf(-(xp[D + j] + hp[D + j])));
          float nn = tanhf(xp[2 * D + j] + r * hp[2 * D + j]);
          float h = sH[m * D + j];
          sH[m * D + j] = (1.0f - z) * nn + z * h;
        }
      }
    }
    // next iteration's top barrier separates these h writes from A-reads
  }
  __syncthreads();

  // ---- LayerNorm statistics (one thread per node; tiny) ----
  if (tid < 16) {
    float mu = 0.0f;
    for (int j = 0; j < D; ++j) mu += sH[tid * D + j];
    mu *= (1.0f / D);
    float var = 0.0f;
    for (int j = 0; j < D; ++j) {
      float d = sH[tid * D + j] - mu;
      var += d * d;
    }
    var *= (1.0f / D);
    sMu[tid] = mu;
    sRs[tid] = rsqrtf(var + LN_EPS);
  }
  __syncthreads();

  // ---- ft = deg==1 ? msgs[:,0] : LN(h)  (stored fp32 into sXp reuse) ----
  {
    int j = tid;
    float g = gamma[j];
    float bt = beta[j];
#pragma unroll
    for (int m = 0; m < 16; ++m) {
      float v;
      if (dg[m] == 1) {
        int src = neigh_idx[(n0 + m) * MAX_DEG + 0];
        v = feat_f32[(size_t)src * D + j];
      } else {
        v = (sH[m * D + j] - sMu[m]) * sRs[m] * g + bt;
      }
      sXp[m * D + j] = v;
    }
  }
  __syncthreads();

  // ---- classifier: out[m][c] = ft[m] . wc[c] + bc[c]  (fp32 scalar) ----
  for (int idx = tid; idx < 16 * N_CLASSES; idx += 128) {
    int m = idx / N_CLASSES;
    int c = idx % N_CLASSES;
    const float* wrow = wc + c * D;
    const float* f = sXp + m * D;
    float acc = bc[c];
#pragma unroll 8
    for (int j = 0; j < D; ++j) acc += f[j] * wrow[j];
    out[(size_t)(n0 + m) * N_CLASSES + c] = acc;
  }
}

// ---------------------------------------------------------------------------
// Host launcher
// ---------------------------------------------------------------------------
extern "C" void kernel_launch(void* const* d_in, const int* in_sizes, int n_in,
                              void* d_out, int out_size, void* d_ws, size_t ws_size,
                              hipStream_t stream) {
  const int* token_id  = (const int*)d_in[0];
  const int* neigh_idx = (const int*)d_in[1];
  const int* deg       = (const int*)d_in[2];
  const float* emb     = (const float*)d_in[3];
  const float* w_ih    = (const float*)d_in[4];
  const float* w_hh    = (const float*)d_in[5];
  const float* b_ih    = (const float*)d_in[6];
  const float* b_hh    = (const float*)d_in[7];
  const float* gamma   = (const float*)d_in[8];
  const float* beta    = (const float*)d_in[9];
  const float* wc      = (const float*)d_in[10];
  const float* bc      = (const float*)d_in[11];
  float* out = (float*)d_out;

  char* ws = (char*)d_ws;
  size_t off = 0;
  float* feat_f32 = (float*)(ws + off);                    off += (size_t)N_SRC * D * sizeof(float);
  unsigned short* feat_bf16 = (unsigned short*)(ws + off); off += (size_t)N_SRC * D * sizeof(unsigned short);
  unsigned short* wih_bf = (unsigned short*)(ws + off);    off += (size_t)3 * D * D * sizeof(unsigned short);
  unsigned short* whh_bf = (unsigned short*)(ws + off);    off += (size_t)3 * D * D * sizeof(unsigned short);
  (void)ws_size; (void)in_sizes; (void)n_in; (void)out_size;

  embed_mean_kernel<<<N_SRC, D, 0, stream>>>(token_id, emb, feat_f32, feat_bf16);
  cvt_bf16_kernel<<<(3 * D * D + 255) / 256, 256, 0, stream>>>(w_ih, wih_bf, 3 * D * D);
  cvt_bf16_kernel<<<(3 * D * D + 255) / 256, 256, 0, stream>>>(w_hh, whh_bf, 3 * D * D);
  gru_fused_kernel<<<N_DST / 16, 128, 0, stream>>>(neigh_idx, deg, feat_f32, feat_bf16,
                                                   wih_bf, whh_bf, b_ih, b_hh,
                                                   gamma, beta, wc, bc, out);
}
